// GCN_attention_v2_14319420965162
// MI455X (gfx1250) — matrix operand order repaired and checked by reading, not compiled, not run on hardware
//
#include <hip/hip_runtime.h>
#include <math.h>

#define NN     4096
#define NFEATC 512
#define NHIDC  128
#define NCLS   16
#define ATTC   64

typedef unsigned short u16;
typedef __attribute__((ext_vector_type(16))) __bf16 bf16x16;
typedef __attribute__((ext_vector_type(16))) u16    u16x16;
typedef __attribute__((ext_vector_type(8)))  u16    u16x8;
typedef __attribute__((ext_vector_type(8)))  float  f32x8;

union BF16Vec { u16x16 u; bf16x16 b; };

__device__ __forceinline__ u16 f2bf(float f) {
  union { float f; unsigned u; } x; x.f = f;
  unsigned r = x.u + 0x7fffu + ((x.u >> 16) & 1u);
  return (u16)(r >> 16);
}
__device__ __forceinline__ float bf2f(u16 h) {
  union { unsigned u; float f; } x; x.u = ((unsigned)h) << 16;
  return x.f;
}

// ---------------------------------------------------------------- converts
__global__ __launch_bounds__(256) void k_f32_to_bf16(
    const float* __restrict__ src, u16* __restrict__ dst, int n) {
  int i = blockIdx.x * blockDim.x + threadIdx.x;
  if (i < n) dst[i] = f2bf(src[i]);
}

// ------------------------------------------------- u_k = Wa_k @ Wagg_k,  c
__global__ __launch_bounds__(256) void k_prep_u(
    const float* __restrict__ Wa1, const float* __restrict__ Wa2,
    const float* __restrict__ Wa3, const float* __restrict__ ba1,
    const float* __restrict__ ba2, const float* __restrict__ ba3,
    const float* __restrict__ Wagg, const float* __restrict__ bagg,
    float* __restrict__ u, float* __restrict__ cvec) {
  int j = blockIdx.x * blockDim.x + threadIdx.x;
  if (j < NN) {
    const float* Wa[3] = {Wa1, Wa2, Wa3};
    for (int k = 0; k < 3; ++k) {
      float a0 = 0.f, a1 = 0.f, a2 = 0.f;
      const float* wr = Wa[k] + (size_t)j * ATTC;
      for (int t = 0; t < ATTC; ++t) {
        float w = wr[t];
        const float* g = Wagg + (size_t)(k * ATTC + t) * 3;
        a0 += w * g[0]; a1 += w * g[1]; a2 += w * g[2];
      }
      u[(size_t)j * 9 + k * 3 + 0] = a0;
      u[(size_t)j * 9 + k * 3 + 1] = a1;
      u[(size_t)j * 9 + k * 3 + 2] = a2;
    }
  }
  if (blockIdx.x == 0 && threadIdx.x == 0) {
    const float* ba[3] = {ba1, ba2, ba3};
    for (int cc = 0; cc < 3; ++cc) {
      float s = bagg[cc];
      for (int k = 0; k < 3; ++k)
        for (int t = 0; t < ATTC; ++t)
          s += ba[k][t] * Wagg[(size_t)(k * ATTC + t) * 3 + cc];
      cvec[cc] = s;
    }
  }
}

// ------------------ z4[i] = sum_j A_k[i,j] u_k[j]; nz = softmax3(z4 + c)
__global__ __launch_bounds__(256) void k_z4_softmax(
    const float* __restrict__ A0, const float* __restrict__ A1,
    const float* __restrict__ A2, const float* __restrict__ u,
    const float* __restrict__ cvec, float* __restrict__ nz,
    float* __restrict__ out_nz) {
  int row  = blockIdx.x * blockDim.y + threadIdx.y;
  int lane = threadIdx.x;
  const float* r0 = A0 + (size_t)row * NN;
  const float* r1 = A1 + (size_t)row * NN;
  const float* r2 = A2 + (size_t)row * NN;
  float acc[9];
#pragma unroll
  for (int t = 0; t < 9; ++t) acc[t] = 0.f;
  for (int j = lane; j < NN; j += 32) {
    float a0 = r0[j], a1 = r1[j], a2 = r2[j];
    const float* uj = u + (size_t)j * 9;
    acc[0] += a0 * uj[0]; acc[1] += a0 * uj[1]; acc[2] += a0 * uj[2];
    acc[3] += a1 * uj[3]; acc[4] += a1 * uj[4]; acc[5] += a1 * uj[5];
    acc[6] += a2 * uj[6]; acc[7] += a2 * uj[7]; acc[8] += a2 * uj[8];
  }
#pragma unroll
  for (int t = 0; t < 9; ++t)
    for (int m = 16; m; m >>= 1) acc[t] += __shfl_xor(acc[t], m, 32);
  if (lane == 0) {
    float z0 = acc[0] + acc[3] + acc[6] + cvec[0];
    float z1 = acc[1] + acc[4] + acc[7] + cvec[1];
    float z2 = acc[2] + acc[5] + acc[8] + cvec[2];
    float m = fmaxf(z0, fmaxf(z1, z2));
    float e0 = __expf(z0 - m), e1 = __expf(z1 - m), e2 = __expf(z2 - m);
    float inv = 1.f / (e0 + e1 + e2);
    float n0 = e0 * inv, n1 = e1 * inv, n2 = e2 * inv;
    nz[row * 3 + 0] = n0; nz[row * 3 + 1] = n1; nz[row * 3 + 2] = n2;
    out_nz[row * 3 + 0] = n0; out_nz[row * 3 + 1] = n1; out_nz[row * 3 + 2] = n2;
  }
}

// ------------- adj[i,j] = sum_k A_k[i,j] * nz[j,k]  (column scaling) -> bf16
__global__ __launch_bounds__(256) void k_adj(
    const float* __restrict__ A0, const float* __restrict__ A1,
    const float* __restrict__ A2, const float* __restrict__ nz,
    u16* __restrict__ adjb) {
  size_t base = ((size_t)blockIdx.x * blockDim.x + threadIdx.x) * 4;
  int j = (int)(base & (size_t)(NN - 1));
  float4 a0 = *(const float4*)(A0 + base);
  float4 a1 = *(const float4*)(A1 + base);
  float4 a2 = *(const float4*)(A2 + base);
  float v0 = a0.x * nz[(j+0)*3+0] + a1.x * nz[(j+0)*3+1] + a2.x * nz[(j+0)*3+2];
  float v1 = a0.y * nz[(j+1)*3+0] + a1.y * nz[(j+1)*3+1] + a2.y * nz[(j+1)*3+2];
  float v2 = a0.z * nz[(j+2)*3+0] + a1.z * nz[(j+2)*3+1] + a2.z * nz[(j+2)*3+2];
  float v3 = a0.w * nz[(j+3)*3+0] + a1.w * nz[(j+3)*3+1] + a2.w * nz[(j+3)*3+2];
  ushort4 s; s.x = f2bf(v0); s.y = f2bf(v1); s.z = f2bf(v2); s.w = f2bf(v3);
  *(ushort4*)(adjb + base) = s;
}

// ------------------------------------------- generic bf16 WMMA GEMM
// C = A[MxK](bf16,row) @ B[KxN](bf16,row); epilogue: +bias, *mask, relu;
// outputs fp32 / bf16 / transposed-bf16 (ldt).
// Block tile 128x64: 8 waves, wave w owns rows [w*16, w*16+16), and computes
// 4 WMMAs across the 64-wide N tile reusing one A operand.
// Requires: M % 128 == 0, K % 32 == 0, N >= 8.
#define BK 32
#define BN 64
#define BM 128

__global__ __launch_bounds__(256) void k_gemm_bf16(
    const u16* __restrict__ A, int lda, const u16* __restrict__ B, int ldb,
    const float* __restrict__ bias, const u16* __restrict__ mask,
    float* __restrict__ outF, u16* __restrict__ outB,
    u16* __restrict__ outBT, int ldt, int M, int N, int K, int relu) {
  __shared__ u16 ldsBT[BN][BK];   // transposed tile: [n][k]

  const int lane = threadIdx.x;
  const int wave = threadIdx.y;
  const int tid  = wave * 32 + lane;
  const int tileM = blockIdx.y * BM + wave * 16;
  const int blkN  = blockIdx.x * BN;
  const int half = lane >> 4;
  const int l15  = lane & 15;

  const u16* arow = A + (size_t)(tileM + l15) * lda;

  const int bk = tid >> 3;        // 0..31 : K within tile
  const int bn = (tid & 7) * 8;   // 0..56 : N within tile
  // Clamped column: LDS columns >= N only feed discarded outputs, so we
  // just keep addresses in-bounds (N >= 8 in all uses here).
  int gcol = blkN + bn;
  int cmax = N - 8;
  if (gcol > cmax) gcol = cmax;
  const u16* bbase = B + gcol;

  f32x8 acc0 = {0.f,0.f,0.f,0.f,0.f,0.f,0.f,0.f};
  f32x8 acc1 = {0.f,0.f,0.f,0.f,0.f,0.f,0.f,0.f};
  f32x8 acc2 = {0.f,0.f,0.f,0.f,0.f,0.f,0.f,0.f};
  f32x8 acc3 = {0.f,0.f,0.f,0.f,0.f,0.f,0.f,0.f};

  for (int k0 = 0; k0 < K; k0 += BK) {
    // cooperative staging of B tile, transposed into LDS (no divergence)
    {
      u16x8 v = *(const u16x8*)(bbase + (size_t)(k0 + bk) * ldb);
#pragma unroll
      for (int e = 0; e < 8; ++e) ldsBT[bn + e][bk] = v[e];
    }
    __syncthreads();

    // A operand: 16-bit A 16x32 layout (half 0: K0..7,K16..23; half 1: +8)
    BF16Vec av;
    {
      const u16* p0 = arow + k0 + half * 8;
      u16x8 lo = *(const u16x8*)p0;
      u16x8 hi = *(const u16x8*)(p0 + 16);
      av.u = __builtin_shufflevector(lo, hi, 0, 1, 2, 3, 4, 5, 6, 7,
                                     8, 9, 10, 11, 12, 13, 14, 15);
    }
    if (k0 + BK < K) __builtin_prefetch(arow + k0 + BK, 0, 1);

    // 4 WMMAs across the N tile, same A operand
#pragma unroll
    for (int nt = 0; nt < 4; ++nt) {
      BF16Vec bv;
      const u16* q = &ldsBT[nt * 16 + l15][half * 8];
      u16x8 lo = *(const u16x8*)q;
      u16x8 hi = *(const u16x8*)(q + 16);
      bv.u = __builtin_shufflevector(lo, hi, 0, 1, 2, 3, 4, 5, 6, 7,
                                     8, 9, 10, 11, 12, 13, 14, 15);
      f32x8& acc = (nt == 0) ? acc0 : (nt == 1) ? acc1 : (nt == 2) ? acc2
                                                                   : acc3;
      acc = __builtin_amdgcn_wmma_f32_16x16x32_bf16(
          false, av.b, false, bv.b, (short)0, acc, false, false);
    }
    __syncthreads();
  }

#pragma unroll
  for (int nt = 0; nt < 4; ++nt) {
    const f32x8& acc = (nt == 0) ? acc0 : (nt == 1) ? acc1 : (nt == 2) ? acc2
                                                                       : acc3;
    int n = blkN + nt * 16 + l15;
    if (n < N) {
      float bval = bias ? bias[n] : 0.f;
#pragma unroll
      for (int r = 0; r < 8; ++r) {
        int m = tileM + half * 8 + r;
        float v = acc[r] + bval;
        if (mask) v *= bf2f(mask[(size_t)m * N + n]);
        if (relu) v = v > 0.f ? v : 0.f;
        size_t idx = (size_t)m * N + n;
        if (outF)  outF[idx] = v;
        if (outB)  outB[idx] = f2bf(v);
        if (outBT) outBT[(size_t)n * ldt + m] = f2bf(v);
      }
    }
  }
}

// --------------------------- row softmax over 4096 (online), fp32 -> bf16
__global__ __launch_bounds__(256) void k_row_softmax_bf16(
    const float* __restrict__ S, u16* __restrict__ P) {
  int row  = blockIdx.x * blockDim.y + threadIdx.y;
  int lane = threadIdx.x;
  const float* r = S + (size_t)row * NN;
  float m = -1e30f, s = 0.f;
  for (int j = lane; j < NN; j += 32) {
    float v = r[j];
    if (v > m) { s = s * __expf(m - v) + 1.f; m = v; }
    else       { s += __expf(v - m); }
  }
  for (int off = 16; off; off >>= 1) {
    float mo = __shfl_xor(m, off, 32);
    float so = __shfl_xor(s, off, 32);
    float M  = fmaxf(m, mo);
    s = s * __expf(m - M) + so * __expf(mo - M);
    m = M;
  }
  float inv = 1.f / s;
  u16* p = P + (size_t)row * NN;
  for (int j = lane; j < NN; j += 32) p[j] = f2bf(__expf(r[j] - m) * inv);
}

// -------------------------------------------------- final 16-wide softmax
__global__ __launch_bounds__(256) void k_softmax16(
    const float* __restrict__ Z, float* __restrict__ out) {
  int i = blockIdx.x * blockDim.x + threadIdx.x;
  if (i >= NN) return;
  const float* z = Z + (size_t)i * NCLS;
  float m = z[0];
#pragma unroll
  for (int c = 1; c < NCLS; ++c) m = fmaxf(m, z[c]);
  float e[NCLS], s = 0.f;
#pragma unroll
  for (int c = 0; c < NCLS; ++c) { e[c] = __expf(z[c] - m); s += e[c]; }
  float inv = 1.f / s;
  float* o = out + (size_t)i * NCLS;
#pragma unroll
  for (int c = 0; c < NCLS; ++c) o[c] = e[c] * inv;
}

// ==========================================================================
extern "C" void kernel_launch(void* const* d_in, const int* in_sizes, int n_in,
                              void* d_out, int out_size, void* d_ws,
                              size_t ws_size, hipStream_t stream) {
  const float* A0   = (const float*)d_in[0];
  const float* A1   = (const float*)d_in[1];
  const float* A2   = (const float*)d_in[2];
  const float* x    = (const float*)d_in[3];
  const float* Wa1  = (const float*)d_in[4];
  const float* ba1  = (const float*)d_in[5];
  const float* Wa2  = (const float*)d_in[6];
  const float* ba2  = (const float*)d_in[7];
  const float* Wa3  = (const float*)d_in[8];
  const float* ba3  = (const float*)d_in[9];
  const float* Wagg = (const float*)d_in[10];
  const float* bagg = (const float*)d_in[11];
  const float* W1   = (const float*)d_in[12];
  const float* b1   = (const float*)d_in[13];
  const float* Wq   = (const float*)d_in[14];
  const float* bq   = (const float*)d_in[15];
  const float* Wk   = (const float*)d_in[16];
  const float* bk   = (const float*)d_in[17];
  const float* Wv   = (const float*)d_in[18];
  const float* bv   = (const float*)d_in[19];
  const float* W2   = (const float*)d_in[20];
  const float* b2   = (const float*)d_in[21];
  float* outF = (float*)d_out;

  // ---- workspace carve (256B aligned)
  char* base = (char*)d_ws;
  size_t off = 0;
  auto carve = [&](size_t bytes) -> char* {
    char* p = base + off;
    off = (off + bytes + 255) & ~(size_t)255;
    return p;
  };
  u16*   adjb  = (u16*)carve((size_t)NN * NN * 2);
  u16*   attnb = (u16*)carve((size_t)NN * NN * 2);
  float* Sf    = (float*)carve((size_t)NN * NN * 4);
  u16*   xb    = (u16*)carve((size_t)NN * NFEATC * 2);
  u16*   W1b   = (u16*)carve((size_t)NFEATC * NHIDC * 2);
  u16*   Wqb   = (u16*)carve((size_t)NHIDC * NHIDC * 2);
  u16*   Wkb   = (u16*)carve((size_t)NHIDC * NHIDC * 2);
  u16*   Wvb   = (u16*)carve((size_t)NHIDC * NHIDC * 2);
  u16*   W2b   = (u16*)carve((size_t)NHIDC * NCLS * 2);
  u16*   xW1b  = (u16*)carve((size_t)NN * NHIDC * 2);
  u16*   hb    = (u16*)carve((size_t)NN * NHIDC * 2);
  u16*   Qb    = (u16*)carve((size_t)NN * NHIDC * 2);
  u16*   KTb   = (u16*)carve((size_t)NHIDC * NN * 2);
  u16*   Vb    = (u16*)carve((size_t)NN * NHIDC * 2);
  u16*   Xtb   = (u16*)carve((size_t)NN * NHIDC * 2);
  u16*   Gb    = (u16*)carve((size_t)NN * NCLS * 2);
  float* ubuf  = (float*)carve((size_t)NN * 9 * 4);
  float* cbuf  = (float*)carve(256);
  float* nzbuf = (float*)carve((size_t)NN * 3 * 4);
  float* zbuf  = (float*)carve((size_t)NN * NCLS * 4);
  (void)ws_size; (void)in_sizes; (void)n_in; (void)out_size;

  auto cvt = [&](const float* s, u16* d, int n) {
    k_f32_to_bf16<<<(n + 255) / 256, 256, 0, stream>>>(s, d, n);
  };
  cvt(x,  xb,  NN * NFEATC);
  cvt(W1, W1b, NFEATC * NHIDC);
  cvt(Wq, Wqb, NHIDC * NHIDC);
  cvt(Wk, Wkb, NHIDC * NHIDC);
  cvt(Wv, Wvb, NHIDC * NHIDC);
  cvt(W2, W2b, NHIDC * NCLS);

  k_prep_u<<<(NN + 255) / 256, 256, 0, stream>>>(Wa1, Wa2, Wa3, ba1, ba2, ba3,
                                                 Wagg, bagg, ubuf, cbuf);
  k_z4_softmax<<<NN / 8, dim3(32, 8), 0, stream>>>(
      A0, A1, A2, ubuf, cbuf, nzbuf, outF + (size_t)NN * NCLS);
  k_adj<<<(int)(((size_t)NN * NN / 4) / 256), 256, 0, stream>>>(
      A0, A1, A2, nzbuf, adjb);

  auto gemm = [&](const u16* A, int lda, const u16* B, int ldb,
                  const float* bias, const u16* mask, float* oF, u16* oB,
                  u16* oBT, int ldt, int M, int Nn, int K, int relu) {
    dim3 grid((Nn + BN - 1) / BN, M / BM), block(32, 8);
    k_gemm_bf16<<<grid, block, 0, stream>>>(A, lda, B, ldb, bias, mask, oF,
                                            oB, oBT, ldt, M, Nn, K, relu);
  };

  // xW1 = x @ W1                                   [N, NHID]
  gemm(xb, NFEATC, W1b, NHIDC, nullptr, nullptr, nullptr, xW1b, nullptr, 0,
       NN, NHIDC, NFEATC, 0);
  // h = relu(adj @ xW1 + b1)
  gemm(adjb, NN, xW1b, NHIDC, b1, nullptr, nullptr, hb, nullptr, 0,
       NN, NHIDC, NN, 1);
  // Qh / Kh(transposed) / Vh
  gemm(hb, NHIDC, Wqb, NHIDC, bq, nullptr, nullptr, Qb, nullptr, 0,
       NN, NHIDC, NHIDC, 0);
  gemm(hb, NHIDC, Wkb, NHIDC, bk, nullptr, nullptr, nullptr, KTb, NN,
       NN, NHIDC, NHIDC, 0);
  gemm(hb, NHIDC, Wvb, NHIDC, bv, nullptr, nullptr, Vb, nullptr, 0,
       NN, NHIDC, NHIDC, 0);
  // A_tilde = adj * (Qh @ Kh^T)                    [N, N] fp32
  gemm(Qb, NHIDC, KTb, NN, nullptr, adjb, Sf, nullptr, nullptr, 0,
       NN, NN, NHIDC, 0);
  // attention = row_softmax(A_tilde); gcn_norm is identity (rows sum to 1)
  k_row_softmax_bf16<<<NN / 8, dim3(32, 8), 0, stream>>>(Sf, attnb);
  // X_tilde = relu(attention @ Vh)
  gemm(attnb, NN, Vb, NHIDC, nullptr, nullptr, nullptr, Xtb, nullptr, 0,
       NN, NHIDC, NN, 1);
  // G = X_tilde @ W2                               [N, 16]
  gemm(Xtb, NHIDC, W2b, NCLS, nullptr, nullptr, nullptr, Gb, nullptr, 0,
       NN, NCLS, NHIDC, 0);
  // z = adj @ G + b2                               [N, 16] fp32
  gemm(adjb, NN, Gb, NCLS, b2, nullptr, zbuf, nullptr, nullptr, 0,
       NN, NCLS, NN, 0);
  // out = softmax(z, axis=1)
  k_softmax16<<<(NN + 255) / 256, 256, 0, stream>>>(zbuf, outF);
}
